// FactorAtt_ConvRelPosEnc_21930103014135
// MI455X (gfx1250) — compile-verified, matrix operand order checked by reading, not run on hardware
//
#include <hip/hip_runtime.h>
#include <hip/hip_bf16.h>

// ---------------------------------------------------------------------------
// FactorAtt_ConvRelPosEnc (CoaT) for MI455X / gfx1250, wave32 + WMMA bf16.
// B=8, H=W=56, N=3136, C=512, heads=8, Ch=64.
// All large GEMMs run as V_WMMA_F32_16X16X32_BF16 with fp32 accumulation.
// ---------------------------------------------------------------------------

typedef __bf16 bhalf;
typedef __attribute__((ext_vector_type(16))) __bf16 v16bf;
typedef __attribute__((ext_vector_type(8)))  __bf16 v8bf;
typedef __attribute__((ext_vector_type(4)))  __bf16 v4bf;
typedef __attribute__((ext_vector_type(8)))  float  v8f;

#define BB   8
#define HS   56
#define WS   56
#define NP   3136      // HS*WS
#define CC   512
#define NH   8
#define CH   64
#define MROWS (BB*NP)  // 25088

#define SMSEG 14       // softmax N-segments (3136/14 = 224)
#define SEGN  224
#define DSEG  7        // dd N-segments (3136/7 = 448 = 14 chunks of 32)
#define DSEGN 448

enum { EPI_NONE = 0, EPI_GELUBN = 1, EPI_BIAS = 2 };

__device__ __forceinline__ float gelu_f(float x) {
  return 0.5f * x * (1.0f + erff(x * 0.70710678118654752f));
}

__device__ __forceinline__ unsigned short bfbits(bhalf x) {
  return __builtin_bit_cast(unsigned short, x);
}

// Build a 16-bf16 WMMA fragment from two contiguous 8-element runs at
// offsets kg and 16+kg (CDNA5 16-bit A/B VGPR layout: lanes 0-15 hold
// K=[0,8)+[16,24), lanes 16-31 hold K=[8,16)+[24,32)).
__device__ __forceinline__ v16bf frag16(const bhalf* p, int kg) {
  v8bf lo = *(const v8bf*)(p + kg);
  v8bf hi = *(const v8bf*)(p + 16 + kg);
  v16bf f;
#pragma unroll
  for (int i = 0; i < 8; ++i) { f[i] = lo[i]; f[i + 8] = hi[i]; }
  return f;
}

// --------------------------- small utility kernels -------------------------

__global__ __launch_bounds__(256) void k_cvt(const float* __restrict__ in,
                                             bhalf* __restrict__ out, int n) {
  int i = blockIdx.x * 256 + threadIdx.x;
  if (i < n) out[i] = (bhalf)in[i];
}

__global__ __launch_bounds__(256) void k_bnprep(
    const float* g0, const float* b0, const float* m0, const float* v0,
    const float* g1, const float* b1, const float* m1, const float* v1,
    float* inv0, float* sh0, float* inv1, float* sh1) {
  int i = blockIdx.x * 256 + threadIdx.x;
  if (i < CC) {
    float iv = g0[i] * rsqrtf(v0[i] + 1e-5f);
    inv0[i] = iv; sh0[i] = b0[i] - m0[i] * iv;
  } else if (i < 2 * CC) {
    int j = i - CC;
    float iv = g1[j] * rsqrtf(v1[j] + 1e-5f);
    inv1[j] = iv; sh1[j] = b1[j] - m1[j] * iv;
  }
}

// --------------------------- depthwise 3x3 conv ----------------------------
template <typename TIN>
__global__ __launch_bounds__(256) void k_dwconv3(const TIN* __restrict__ in,
                                                 const float* __restrict__ w,
                                                 bhalf* __restrict__ out) {
  int idx = blockIdx.x * 256 + threadIdx.x;
  int c = idx & (CC - 1);
  int rest = idx >> 9;
  int n = rest % NP;
  int b = rest / NP;
  int y = n / WS, x = n % WS;
  const float* wc = w + c * 9;
  float s = 0.f;
#pragma unroll
  for (int dy = -1; dy <= 1; ++dy) {
    int yy = y + dy;
    if ((unsigned)yy >= (unsigned)HS) continue;
#pragma unroll
    for (int dx = -1; dx <= 1; ++dx) {
      int xx = x + dx;
      if ((unsigned)xx >= (unsigned)WS) continue;
      s += (float)in[((size_t)(b * NP + yy * WS + xx) << 9) + c] *
           wc[(dy + 1) * 3 + (dx + 1)];
    }
  }
  out[idx] = (bhalf)s;
}

// --------------------------- CRPE grouped depthwise conv -------------------
__global__ __launch_bounds__(256) void k_crpe(const bhalf* __restrict__ v,
    const float* __restrict__ w3, const float* __restrict__ b3,
    const float* __restrict__ w5, const float* __restrict__ b5,
    const float* __restrict__ w7, const float* __restrict__ b7,
    bhalf* __restrict__ out) {
  int idx = blockIdx.x * 256 + threadIdx.x;
  int c = idx & (CC - 1);
  int rest = idx >> 9;
  int n = rest % NP;
  int b = rest / NP;
  int y = n / WS, x = n % WS;
  const float* wc; float bias; int ks;
  if (c < 128)      { wc = w3 + c * 9;          bias = b3[c];       ks = 3; }
  else if (c < 320) { wc = w5 + (c - 128) * 25; bias = b5[c - 128]; ks = 5; }
  else              { wc = w7 + (c - 320) * 49; bias = b7[c - 320]; ks = 7; }
  int p = ks >> 1;
  float s = bias;
  for (int dy = 0; dy < ks; ++dy) {
    int yy = y + dy - p;
    if ((unsigned)yy >= (unsigned)HS) continue;
    for (int dx = 0; dx < ks; ++dx) {
      int xx = x + dx - p;
      if ((unsigned)xx >= (unsigned)WS) continue;
      s += (float)v[((size_t)(b * NP + yy * WS + xx) << 9) + c] * wc[dy * ks + dx];
    }
  }
  out[idx] = (bhalf)s;
}

// --------------------------- main WMMA GEMM --------------------------------
// out[m, n] = sum_k A[m,k] * W[n,k], M x 512 x 512.
// Each wave owns a 32x64 macro-tile: 2 A-frags x 4 B-frags -> 8 WMMAs per
// K-chunk. Register-direct fragment loads (the whole inter-kernel working set
// is L2-resident: 192MB L2); global_prefetch hides L2-hit latency.
template <typename TA, int EPI>
__global__ __launch_bounds__(256) void k_gemm(const TA* __restrict__ A,
                                              const bhalf* __restrict__ W,
                                              void* __restrict__ outp,
                                              const float* __restrict__ bnInv,
                                              const float* __restrict__ bnShift,
                                              const float* __restrict__ bias,
                                              int M) {
  constexpr int K = CC, COLS = CC;
  const int lane = threadIdx.x & 31;
  const int wid = threadIdx.x >> 5;
  const int wtile = blockIdx.x * 8 + wid;       // (M/32)*(COLS/64) waves
  const int tn4 = wtile & (COLS / 64 - 1);      // 8 column macro-tiles
  const int tm = wtile >> 3;                    // 32-row macro-tile
  if (tm * 32 >= M) return;
  const int l15 = lane & 15;
  const int kg = (lane >> 4) << 3;
  const TA* arow0 = A + (size_t)(tm * 32 + l15) * K;
  const TA* arow1 = arow0 + (size_t)16 * K;
  const bhalf* wrow = W + (size_t)(tn4 * 64 + l15) * K;
  v8f acc[2][4] = {};
#pragma unroll 2
  for (int k0 = 0; k0 < K; k0 += 32) {
    if (k0 + 64 < K) {  // speculative prefetch 2 chunks ahead (global_prefetch)
      __builtin_prefetch(arow0 + k0 + 64, 0, 1);
      __builtin_prefetch(arow1 + k0 + 64, 0, 1);
    }
    v16bf a0, a1;
    if constexpr (__is_same(TA, float)) {
      v8f lo0 = *(const v8f*)(arow0 + k0 + kg);
      v8f hi0 = *(const v8f*)(arow0 + k0 + 16 + kg);
      v8f lo1 = *(const v8f*)(arow1 + k0 + kg);
      v8f hi1 = *(const v8f*)(arow1 + k0 + 16 + kg);
#pragma unroll
      for (int i = 0; i < 8; ++i) {
        a0[i] = (bhalf)lo0[i]; a0[i + 8] = (bhalf)hi0[i];
        a1[i] = (bhalf)lo1[i]; a1[i + 8] = (bhalf)hi1[i];
      }
    } else {
      a0 = frag16(arow0 + k0, kg);
      a1 = frag16(arow1 + k0, kg);
    }
#pragma unroll
    for (int j = 0; j < 4; ++j) {
      const bhalf* wj = wrow + (size_t)j * 16 * K + k0;
      if (k0 + 64 < K) __builtin_prefetch(wj + 64, 0, 1);
      v16bf bf_ = frag16(wj, kg);
      acc[0][j] = __builtin_amdgcn_wmma_f32_16x16x32_bf16(false, a0, false, bf_,
                                                          (short)0, acc[0][j], false, false);
      acc[1][j] = __builtin_amdgcn_wmma_f32_16x16x32_bf16(false, a1, false, bf_,
                                                          (short)0, acc[1][j], false, false);
    }
  }
  const int rb = (lane >> 4) << 3;
#pragma unroll
  for (int i = 0; i < 2; ++i) {
#pragma unroll
    for (int j = 0; j < 4; ++j) {
      const int col = tn4 * 64 + j * 16 + l15;
#pragma unroll
      for (int r = 0; r < 8; ++r) {
        int m = tm * 32 + i * 16 + rb + r;
        float vv = acc[i][j][r];
        if constexpr (EPI == EPI_GELUBN) {
          vv = gelu_f(vv) * bnInv[col] + bnShift[col];
          ((bhalf*)outp)[(size_t)m * COLS + col] = (bhalf)vv;
        } else if constexpr (EPI == EPI_BIAS) {
          ((float*)outp)[(size_t)m * COLS + col] = vv + bias[col];
        } else {
          ((bhalf*)outp)[(size_t)m * COLS + col] = (bhalf)vv;
        }
      }
    }
  }
}

// --------------------------- column softmax over N (3 phases) --------------
__global__ __launch_bounds__(256) void k_sm_part(const bhalf* __restrict__ kk,
                                                 float* __restrict__ pmax,
                                                 float* __restrict__ psum) {
  int col = blockIdx.x * 256 + threadIdx.x;   // 0..B*C-1
  int seg = blockIdx.y;                        // 0..SMSEG-1
  int c = col & (CC - 1);
  int b = col >> 9;
  const bhalf* p = kk + (size_t)b * NP * CC + (size_t)seg * SEGN * CC + c;
  float mx = -1e30f;
  for (int n = 0; n < SEGN; ++n) mx = fmaxf(mx, (float)p[(size_t)n * CC]);
  float s = 0.f;
  for (int n = 0; n < SEGN; ++n) s += __expf((float)p[(size_t)n * CC] - mx);
  pmax[seg * (BB * CC) + col] = mx;
  psum[seg * (BB * CC) + col] = s;
}

__global__ __launch_bounds__(256) void k_sm_comb(const float* __restrict__ pmax,
                                                 const float* __restrict__ psum,
                                                 float* __restrict__ gmax,
                                                 float* __restrict__ ginv) {
  int col = blockIdx.x * 256 + threadIdx.x;
  float mx = -1e30f;
#pragma unroll
  for (int s = 0; s < SMSEG; ++s) mx = fmaxf(mx, pmax[s * (BB * CC) + col]);
  float sum = 0.f;
#pragma unroll
  for (int s = 0; s < SMSEG; ++s)
    sum += psum[s * (BB * CC) + col] * __expf(pmax[s * (BB * CC) + col] - mx);
  gmax[col] = mx;
  ginv[col] = 1.0f / sum;
}

__global__ __launch_bounds__(256) void k_sm_norm(bhalf* __restrict__ kk,
                                                 const float* __restrict__ gmax,
                                                 const float* __restrict__ ginv) {
  int idx = blockIdx.x * 256 + threadIdx.x;
  int c = idx & (CC - 1);
  int b = idx / (NP * CC);
  int col = (b << 9) + c;
  kk[idx] = (bhalf)(__expf((float)kk[idx] - gmax[col]) * ginv[col]);
}

// --------------------------- dd = k_sm^T @ v per (b,h), N-split ------------
// Grid: DSEG * B*NH blocks; each accumulates a 448-row partial into its own
// dd slice; k_combine sums the DSEG partials. Staging: each thread loads an
// n-pair (8B per row) and emits packed ds_store_b32 into the transposed
// [c][n] tile (n even -> 4B-aligned; row stride 80B).
__global__ __launch_bounds__(256) void k_dd(const bhalf* __restrict__ ksm,
                                            const bhalf* __restrict__ v,
                                            float* __restrict__ dd) {
  __shared__ bhalf skt[64][40];   // [c][n], padded row (80B, 16B-divisible)
  __shared__ bhalf svt[64][40];
  const int bh = blockIdx.x & 63;
  const int seg = blockIdx.x >> 6;
  const int b = bh >> 3;
  const int hh = bh & 7;
  const int t = threadIdx.x;
  const int lane = t & 31, wid = t >> 5;
  const int l15 = lane & 15;
  const int kg = (lane >> 4) << 3;
  const int tm0 = wid >> 2, tn0 = wid & 3;
  const int tm1 = (wid + 8) >> 2, tn1 = (wid + 8) & 3;
  const int npair = (t >> 4) << 1;     // 0,2,..,30 (even n)
  const int cq = (t & 15) << 2;        // 0..60 step 4
  v8f acc0 = {}, acc1 = {};
  const int nbeg = seg * DSEGN;
  for (int n0 = nbeg; n0 < nbeg + DSEGN; n0 += 32) {
    size_t g0 = ((size_t)(b * NP + n0 + npair) << 9) + (size_t)hh * 64 + cq;
    size_t g1 = g0 + CC;               // next n row
    v4bf ka = *(const v4bf*)(ksm + g0);
    v4bf kb = *(const v4bf*)(ksm + g1);
    v4bf va = *(const v4bf*)(v + g0);
    v4bf vb = *(const v4bf*)(v + g1);
#pragma unroll
    for (int i = 0; i < 4; ++i) {
      unsigned int kp = (unsigned int)bfbits(ka[i]) |
                        ((unsigned int)bfbits(kb[i]) << 16);
      unsigned int vp = (unsigned int)bfbits(va[i]) |
                        ((unsigned int)bfbits(vb[i]) << 16);
      *(unsigned int*)&skt[cq + i][npair] = kp;
      *(unsigned int*)&svt[cq + i][npair] = vp;
    }
    __syncthreads();
    {
      v16bf a0 = frag16(&skt[tm0 * 16 + l15][0], kg);
      v16bf b0 = frag16(&svt[tn0 * 16 + l15][0], kg);
      acc0 = __builtin_amdgcn_wmma_f32_16x16x32_bf16(false, a0, false, b0,
                                                     (short)0, acc0, false, false);
      v16bf a1 = frag16(&skt[tm1 * 16 + l15][0], kg);
      v16bf b1 = frag16(&svt[tn1 * 16 + l15][0], kg);
      acc1 = __builtin_amdgcn_wmma_f32_16x16x32_bf16(false, a1, false, b1,
                                                     (short)0, acc1, false, false);
    }
    __syncthreads();
  }
  float* base = dd + ((size_t)seg * 64 + bh) * 4096;
  const int rb = (lane >> 4) << 3;
#pragma unroll
  for (int r = 0; r < 8; ++r) {
    base[(tm0 * 16 + rb + r) * 64 + tn0 * 16 + l15] = acc0[r];
    base[(tm1 * 16 + rb + r) * 64 + tn1 * 16 + l15] = acc1[r];
  }
}

// --------------------------- combine: q@dd * scale + q.*conv_v -------------
__global__ __launch_bounds__(256) void k_combine(const bhalf* __restrict__ q,
                                                 const float* __restrict__ dd,
                                                 const bhalf* __restrict__ convv,
                                                 bhalf* __restrict__ z) {
  __shared__ bhalf ddT[64][72];   // [d][c], 144B rows (16B-divisible)
  const int bid = blockIdx.x;
  const int b = bid / (NH * 25);
  const int rem = bid % (NH * 25);
  const int hh = rem / 25;
  const int mblk = rem % 25;
  const int bh = b * 8 + hh;
  {
    int k = threadIdx.x >> 2;            // 0..63 (c index)
    int d0 = (threadIdx.x & 3) << 4;     // 0,16,32,48
#pragma unroll
    for (int i = 0; i < 16; ++i) {
      float s = 0.f;
#pragma unroll
      for (int g = 0; g < DSEG; ++g)
        s += dd[((size_t)g * 64 + bh) * 4096 + k * 64 + d0 + i];
      ddT[d0 + i][k] = (bhalf)s;
    }
  }
  __syncthreads();
  const int lane = threadIdx.x & 31, wid = threadIdx.x >> 5;
  const int tileM = mblk * 8 + wid;
  if (tileM >= NP / 16) return;          // uniform per wave
  const int l15 = lane & 15;
  const int kg = (lane >> 4) << 3;
  const int m = tileM * 16 + l15;
  const bhalf* qrow = q + ((size_t)(b * NP) + m) * CC + hh * 64;
  v16bf aA = frag16(qrow, kg);           // K chunk [0,32)
  v16bf aB = frag16(qrow + 32, kg);      // K chunk [32,64)
  v8f acc[4] = {};
#pragma unroll
  for (int j = 0; j < 4; ++j) {
    const bhalf* brow = &ddT[j * 16 + l15][0];
    v16bf b0 = frag16(brow, kg);
    v16bf b1 = frag16(brow + 32, kg);
    acc[j] = __builtin_amdgcn_wmma_f32_16x16x32_bf16(false, aA, false, b0,
                                                     (short)0, acc[j], false, false);
    acc[j] = __builtin_amdgcn_wmma_f32_16x16x32_bf16(false, aB, false, b1,
                                                     (short)0, acc[j], false, false);
  }
  const int rb = (lane >> 4) << 3;
#pragma unroll
  for (int j = 0; j < 4; ++j) {
#pragma unroll
    for (int r = 0; r < 8; ++r) {
      int mm = tileM * 16 + rb + r;
      int col = hh * 64 + j * 16 + l15;
      size_t o = ((size_t)(b * NP) + mm) * CC + col;
      float qv = (float)q[o];
      float cv = (float)convv[o];
      z[o] = (bhalf)(0.125f * acc[j][r] + qv * cv);   // scale = Ch^-0.5
    }
  }
}

// ---------------------------------------------------------------------------

extern "C" void kernel_launch(void* const* d_in, const int* in_sizes, int n_in,
                              void* d_out, int out_size, void* d_ws, size_t ws_size,
                              hipStream_t stream) {
  (void)in_sizes; (void)n_in; (void)out_size; (void)ws_size;
  const float* x      = (const float*)d_in[0];
  const float* wq     = (const float*)d_in[1];
  const float* wk     = (const float*)d_in[2];
  const float* wv     = (const float*)d_in[3];
  const float* wproj  = (const float*)d_in[4];
  const float* bproj  = (const float*)d_in[5];
  const float* pe0_dw = (const float*)d_in[6];
  const float* pe0_pw = (const float*)d_in[7];
  const float* pe0_g  = (const float*)d_in[8];
  const float* pe0_b  = (const float*)d_in[9];
  const float* pe0_m  = (const float*)d_in[10];
  const float* pe0_v  = (const float*)d_in[11];
  const float* pe1_dw = (const float*)d_in[12];
  const float* pe1_pw = (const float*)d_in[13];
  const float* pe1_g  = (const float*)d_in[14];
  const float* pe1_b  = (const float*)d_in[15];
  const float* pe1_m  = (const float*)d_in[16];
  const float* pe1_v  = (const float*)d_in[17];
  const float* cw3 = (const float*)d_in[18];
  const float* cb3 = (const float*)d_in[19];
  const float* cw5 = (const float*)d_in[20];
  const float* cb5 = (const float*)d_in[21];
  const float* cw7 = (const float*)d_in[22];
  const float* cb7 = (const float*)d_in[23];

  // ---- workspace layout (all sizes 256B-multiples) ----
  char* ws = (char*)d_ws;
  const size_t WSZ = (size_t)CC * CC * sizeof(bhalf);       // 524288
  bhalf* wqb    = (bhalf*)(ws + 0 * WSZ);
  bhalf* wkb    = (bhalf*)(ws + 1 * WSZ);
  bhalf* wvb    = (bhalf*)(ws + 2 * WSZ);
  bhalf* wprojb = (bhalf*)(ws + 3 * WSZ);
  bhalf* pw0b   = (bhalf*)(ws + 4 * WSZ);
  bhalf* pw1b   = (bhalf*)(ws + 5 * WSZ);
  size_t off = 6 * WSZ;
  float* inv0 = (float*)(ws + off); off += 2048;
  float* sh0  = (float*)(ws + off); off += 2048;
  float* inv1 = (float*)(ws + off); off += 2048;
  float* sh1  = (float*)(ws + off); off += 2048;
  float* ddp  = (float*)(ws + off); off += (size_t)DSEG * 64 * 4096 * 4;   // 7.34MB
  float* pmax = (float*)(ws + off); off += (size_t)SMSEG * BB * CC * 4;
  float* psum = (float*)(ws + off); off += (size_t)SMSEG * BB * CC * 4;
  float* gmax = (float*)(ws + off); off += (size_t)BB * CC * 4;
  float* ginv = (float*)(ws + off); off += (size_t)BB * CC * 4;
  const size_t SLOT = (size_t)MROWS * CC * sizeof(bhalf);   // 25690112
  bhalf* tB  = (bhalf*)(ws + off); off += SLOT;  // dwconv out / conv_v
  bhalf* a1B = (bhalf*)(ws + off); off += SLOT;  // a1 / z (reused)
  bhalf* a2B = (bhalf*)(ws + off); off += SLOT;
  bhalf* qB  = (bhalf*)(ws + off); off += SLOT;
  bhalf* kB  = (bhalf*)(ws + off); off += SLOT;
  bhalf* vB  = (bhalf*)(ws + off); off += SLOT;

  const int GCVT  = (CC * CC + 255) / 256;            // 1024
  const int GELT  = (MROWS * CC) / 256;               // 50176
  const int GGEMM = (MROWS / 32) * (CC / 64) / 8;     // 784 blocks (6272 waves)

  // weights -> bf16; fold BN
  k_cvt<<<GCVT, 256, 0, stream>>>(wq, wqb, CC * CC);
  k_cvt<<<GCVT, 256, 0, stream>>>(wk, wkb, CC * CC);
  k_cvt<<<GCVT, 256, 0, stream>>>(wv, wvb, CC * CC);
  k_cvt<<<GCVT, 256, 0, stream>>>(wproj, wprojb, CC * CC);
  k_cvt<<<GCVT, 256, 0, stream>>>(pe0_pw, pw0b, CC * CC);
  k_cvt<<<GCVT, 256, 0, stream>>>(pe1_pw, pw1b, CC * CC);
  k_bnprep<<<4, 256, 0, stream>>>(pe0_g, pe0_b, pe0_m, pe0_v,
                                  pe1_g, pe1_b, pe1_m, pe1_v,
                                  inv0, sh0, inv1, sh1);

  // patch embeds: dw3x3 -> pw GEMM(+GELU+BN)
  k_dwconv3<float><<<GELT, 256, 0, stream>>>(x, pe0_dw, tB);
  k_gemm<bhalf, EPI_GELUBN><<<GGEMM, 256, 0, stream>>>(tB, pw0b, a1B, inv0, sh0, nullptr, MROWS);
  k_dwconv3<bhalf><<<GELT, 256, 0, stream>>>(a1B, pe1_dw, tB);
  k_gemm<bhalf, EPI_GELUBN><<<GGEMM, 256, 0, stream>>>(tB, pw1b, a2B, inv1, sh1, nullptr, MROWS);

  // q/k/v projections
  k_gemm<float, EPI_NONE><<<GGEMM, 256, 0, stream>>>(x, wqb, qB, nullptr, nullptr, nullptr, MROWS);
  k_gemm<bhalf, EPI_NONE><<<GGEMM, 256, 0, stream>>>(a1B, wkb, kB, nullptr, nullptr, nullptr, MROWS);
  k_gemm<bhalf, EPI_NONE><<<GGEMM, 256, 0, stream>>>(a2B, wvb, vB, nullptr, nullptr, nullptr, MROWS);

  // softmax over keys (axis N): partials -> combine -> normalize
  {
    dim3 gp((BB * CC) / 256, SMSEG);
    k_sm_part<<<gp, 256, 0, stream>>>(kB, pmax, psum);
    k_sm_comb<<<(BB * CC) / 256, 256, 0, stream>>>(pmax, psum, gmax, ginv);
    k_sm_norm<<<GELT, 256, 0, stream>>>(kB, gmax, ginv);
  }

  // dd = k_sm^T @ v per (b,h), split over N into DSEG partials
  k_dd<<<DSEG * BB * NH, 256, 0, stream>>>(kB, vB, ddp);

  // CRPE depthwise conv on v -> conv_v (reuse tB)
  k_crpe<<<GELT, 256, 0, stream>>>(vB, cw3, cb3, cw5, cb5, cw7, cb7, tB);

  // z = scale * q@dd + q .* conv_v  (sums DSEG dd partials; reuse a1B as z)
  k_combine<<<BB * NH * 25, 256, 0, stream>>>(qB, ddp, tB, a1B);

  // out = z @ wproj^T + bproj   (fp32 output)
  k_gemm<bhalf, EPI_BIAS><<<GGEMM, 256, 0, stream>>>(a1B, wprojb, (float*)d_out,
                                                     nullptr, nullptr, bproj, MROWS);
}
